// minGRU_38482906972365
// MI455X (gfx1250) — matrix operand order repaired and checked by reading, not compiled, not run on hardware
//
#include <hip/hip_runtime.h>
#include <hip/hip_bf16.h>
#include <math.h>

// ---------------- problem constants (from reference) ----------------
#define BSZ   4
#define SLEN  4096
#define DM    1024
#define MTOT  (BSZ * SLEN)      // 16384 GEMM rows
// ---------------- GEMM tiling ----------------
#define BM 128                  // rows per block (8 waves x 16)
#define BN 64                   // hidden cols per block (gate cols mirrored at +DM)
#define BK 32                   // K per stage (one WMMA K)
#define NSTAGE (DM / BK)        // 32
#define LDS_STRIDE 40           // bf16 per LDS row (BK + 8 pad -> 80B = 20 banks)
// ---------------- scan chunking ----------------
#define CHUNK  128
#define NCHUNK (SLEN / CHUNK)   // 32

typedef __bf16 bf16_t;
typedef bf16_t bf16x16 __attribute__((ext_vector_type(16)));
typedef float  f32x8   __attribute__((ext_vector_type(8)));
typedef unsigned int u32x4 __attribute__((ext_vector_type(4)));
typedef int i32x4 __attribute__((ext_vector_type(4)));
typedef int i32x8 __attribute__((ext_vector_type(8)));

union FragB16 { uint4 q[2]; bf16x16 v; };

#if defined(__HIP_DEVICE_COMPILE__) && defined(__has_builtin)
#if __has_builtin(__builtin_amdgcn_tensor_load_to_lds)
#define USE_TDM 1
#endif
#endif

// fp32 -> bf16 RNE (inputs are finite gaussian products; skip NaN path)
static __device__ inline unsigned f2bf(float f) {
    unsigned u = __float_as_uint(f);
    return (u + 0x7FFFu + ((u >> 16) & 1u)) >> 16;
}

static __device__ inline float softplusf(float x) {
    return fmaxf(x, 0.0f) + log1pf(__expf(-fabsf(x)));
}

static __device__ inline float logaddexpf(float a, float b) {
    float mx = fmaxf(a, b), mn = fminf(a, b);
    if (mx == -INFINITY) return -INFINITY;
    return mx + log1pf(__expf(mn - mx));
}

// =====================================================================
// Kernel 0: one-shot fp32 -> bf16 of x and W (both stay L2-resident:
// 36 MB total vs 192 MB L2), so the GEMM staging path needs no VALU.
// =====================================================================
__global__ __launch_bounds__(256)
void cvt_f32_to_bf16(const float* __restrict__ src, unsigned short* __restrict__ dst)
{
    const int i = blockIdx.x * 256 + threadIdx.x;       // 8 elements / thread
    const float4 a = ((const float4*)src)[2 * i];
    const float4 b = ((const float4*)src)[2 * i + 1];
    uint4 o;
    o.x = f2bf(a.x) | (f2bf(a.y) << 16);
    o.y = f2bf(a.z) | (f2bf(a.w) << 16);
    o.z = f2bf(b.x) | (f2bf(b.y) << 16);
    o.w = f2bf(b.z) | (f2bf(b.w) << 16);
    ((uint4*)dst)[i] = o;
}

// =====================================================================
// TDM descriptor issue: 2D bf16 tile -> LDS. pad_interval=3 (16 DWORDs
// = one 64B row) + pad_amount=3 (4 DWORDs = 16B) hardware-reproduces the
// 80B / 20-bank padded LDS row stride. type=2 ("image"), data_size=1 (2B).
// =====================================================================
#ifdef USE_TDM
static __device__ inline void tdm_tile_2d(unsigned lds_off, const unsigned short* gptr,
                                          unsigned tile_h, unsigned tensor_h,
                                          unsigned stride_elems)
{
    unsigned long long ga = (unsigned long long)(uintptr_t)gptr;
    u32x4 g0;
    g0[0] = 1u;                                            // count=1, user descriptor
    g0[1] = lds_off;                                       // lds_addr (bytes)
    g0[2] = (unsigned)(ga & 0xFFFFFFFFu);                  // global_addr[31:0]
    g0[3] = (unsigned)((ga >> 32) & 0x01FFFFFFu) | (2u << 30); // addr[56:32] | type=2
    i32x8 g1;
    g1[0] = (1 << 16) | (1 << 20) | (3 << 22) | (3 << 25); // data_size=2B, pad_en, ivl=64B, amt=16B
    g1[1] = (int)((stride_elems & 0xFFFFu) << 16);         // tensor_dim0 lo (== row length)
    g1[2] = (int)(((stride_elems >> 16) & 0xFFFFu) | ((tensor_h & 0xFFFFu) << 16));
    g1[3] = (int)(((tensor_h >> 16) & 0xFFFFu) | ((unsigned)BK << 16)); // tile_dim0 = BK
    g1[4] = (int)(tile_h & 0xFFFFu);                       // tile_dim1 (tile_dim2 = 0)
    g1[5] = (int)stride_elems;                             // tensor_dim0_stride lo
    g1[6] = 0;
    g1[7] = 0;
    i32x4 z4 = {0, 0, 0, 0};
#if defined(__clang_major__) && __clang_major__ >= 23
    i32x8 z8 = {0, 0, 0, 0, 0, 0, 0, 0};
    __builtin_amdgcn_tensor_load_to_lds(g0, g1, z4, z4, z8, 0);
#else
    __builtin_amdgcn_tensor_load_to_lds(g0, g1, z4, z4, 0);
#endif
}
#endif

// stage a (BM x BK) A tile + (2*BN x BK) B tile (hidden rows ++ gate rows)
static __device__ inline void stage_load(const unsigned short* __restrict__ xb,
                                         const unsigned short* __restrict__ wb,
                                         unsigned short* lA, unsigned short* lB,
                                         int m0, int n0, int kk)
{
#ifdef USE_TDM
    if (threadIdx.x < 32) {          // one wave issues DMA; EXEC ignored by TDM
        tdm_tile_2d((unsigned)(uintptr_t)lA,
                    xb + (size_t)m0 * DM + kk, BM, MTOT, DM);
        tdm_tile_2d((unsigned)(uintptr_t)lB,
                    wb + (size_t)n0 * DM + kk, BN, 2 * DM, DM);
        tdm_tile_2d((unsigned)(uintptr_t)(lB + BN * LDS_STRIDE),
                    wb + (size_t)(DM + n0) * DM + kk, BN, 2 * DM, DM);
    }
#else
    const int tid = threadIdx.x;
    #pragma unroll
    for (int p = 0; p < 2; ++p) {                          // A: 128 rows x 4 16B chunks
        int idx = p * 256 + tid; int row = idx >> 2, cg = idx & 3;
        *(uint4*)&lA[row * LDS_STRIDE + cg * 8] =
            *(const uint4*)(xb + (size_t)(m0 + row) * DM + kk + cg * 8);
    }
    #pragma unroll
    for (int p = 0; p < 2; ++p) {                          // B: 128 rows (64 hidden + 64 gate)
        int idx = p * 256 + tid; int row = idx >> 2, cg = idx & 3;
        int e = (row < BN) ? (n0 + row) : (DM + n0 + (row - BN));
        *(uint4*)&lB[row * LDS_STRIDE + cg * 8] =
            *(const uint4*)(wb + (size_t)e * DM + kk + cg * 8);
    }
#endif
}

// =====================================================================
// Kernel 1: hg = x * W^T via v_wmma_f32_16x16x32_bf16.
// TDM double-buffered pipeline: wait tensorcnt -> barrier -> issue next
// stage's DMA -> 8 WMMAs on current buffer. Fused log-space epilogue.
// =====================================================================
__global__ __launch_bounds__(256)
void mingru_gemm_epilogue(const unsigned short* __restrict__ xb,
                          const unsigned short* __restrict__ wb,
                          float* __restrict__ logc,
                          float* __restrict__ logv)
{
    __shared__ __align__(16) unsigned short lA[2][BM * LDS_STRIDE];
    __shared__ __align__(16) unsigned short lB[2][2 * BN * LDS_STRIDE];

    const int tid  = threadIdx.x;
    const int wave = tid >> 5;        // 0..7 -> M subtile
    const int lane = tid & 31;
    const int l16  = lane & 15;
    const int hsel = lane >> 4;       // K-half select per ISA fragment layout
    const int m0   = blockIdx.x * BM;
    const int n0   = blockIdx.y * BN;

    // acc[0..3] = hidden subtiles n0+{0,16,32,48}; acc[4..7] = paired gate subtiles
    f32x8 acc[8] = {};

    stage_load(xb, wb, &lA[0][0], &lB[0][0], m0, n0, 0);   // prologue

    for (int stage = 0; stage < NSTAGE; ++stage) {
        const int buf = stage & 1;
#ifdef USE_TDM
        if (threadIdx.x < 32) __builtin_amdgcn_s_wait_tensorcnt(0);
#endif
        __syncthreads();                                   // stage data visible to all waves
        if (stage + 1 < NSTAGE)                            // DMA-prefetch next stage
            stage_load(xb, wb, &lA[buf ^ 1][0], &lB[buf ^ 1][0],
                       m0, n0, (stage + 1) * BK);

        // A fragment (16x32): lane<16 K{0..7,16..23}, lane>=16 K{8..15,24..31}
        FragB16 fa;
        const char* arow = (const char*)&lA[buf][(wave * 16 + l16) * LDS_STRIDE];
        fa.q[0] = *(const uint4*)(arow + hsel * 16);
        fa.q[1] = *(const uint4*)(arow + 32 + hsel * 16);

        // preload ALL B fragments, then a clean back-to-back WMMA chain
        FragB16 fb[8];
        #pragma unroll
        for (int j = 0; j < 8; ++j) {
            const int brow = (j < 4) ? (j * 16 + l16) : (BN + (j - 4) * 16 + l16);
            const char* bp = (const char*)&lB[buf][brow * LDS_STRIDE];
            fb[j].q[0] = *(const uint4*)(bp + hsel * 16);
            fb[j].q[1] = *(const uint4*)(bp + 32 + hsel * 16);
        }
        #pragma unroll
        for (int j = 0; j < 8; ++j)
            acc[j] = __builtin_amdgcn_wmma_f32_16x16x32_bf16(
                false, fa.v, false, fb[j].v, (short)0, acc[j], false, false);

        __syncthreads();                                   // done reading buf
    }

    // fused epilogue: C/D layout VGPR r, lane L -> M = r + 8*(L>>4), N = L&15
    #pragma unroll
    for (int j = 0; j < 4; ++j) {
        const int col = n0 + j * 16 + l16;
        #pragma unroll
        for (int r = 0; r < 8; ++r) {
            const int m = m0 + wave * 16 + r + 8 * hsel;
            float h = acc[j][r];
            float g = acc[4 + j][r];
            float lc = -softplusf(g);                        // log(1-z)
            float lz = -softplusf(-g);                       // log z
            float lg = (h >= 0.0f) ? __logf(h + 0.5f) : -softplusf(-h);
            size_t o = (size_t)m * DM + col;
            logc[o] = lc;
            logv[o] = lz + lg;
        }
    }
}

// =====================================================================
// Kernel 2: per-chunk local scan summaries (131072 independent streams).
// =====================================================================
__global__ __launch_bounds__(256)
void mingru_chunk_scan(const float* __restrict__ logc,
                       const float* __restrict__ logv,
                       float* __restrict__ Ac, float* __restrict__ Vc)
{
    const int gid = blockIdx.x * 256 + threadIdx.x;
    const int d = gid & (DM - 1);
    const int c = (gid / DM) & (NCHUNK - 1);
    const int b = gid / (DM * NCHUNK);
    const size_t base = ((size_t)b * SLEN + (size_t)c * CHUNK) * DM + d;

    float la = 0.0f, m = -INFINITY, s = 0.0f;
    for (int t = 0; t < CHUNK; ++t) {
        __builtin_prefetch(&logc[base + (size_t)(t + 4) * DM], 0, 1);
        __builtin_prefetch(&logv[base + (size_t)(t + 4) * DM], 0, 1);
        const float lc = logc[base + (size_t)t * DM];
        const float lv = logv[base + (size_t)t * DM];
        la += lc;
        const float v = lv - la;
        if (v > m) { s = s * __expf(m - v) + 1.0f; m = v; }
        else       { s += __expf(v - m); }
    }
    Ac[gid] = la;
    Vc[gid] = m + __logf(s);
}

// =====================================================================
// Kernel 3: exclusive prefix over 32 chunk summaries per channel.
// =====================================================================
__global__ __launch_bounds__(256)
void mingru_chunk_prefix(const float* __restrict__ Ac,
                         const float* __restrict__ Vc,
                         float* __restrict__ Wc)
{
    const int gid = blockIdx.x * 256 + threadIdx.x;     // 0 .. B*D-1
    const int d = gid & (DM - 1);
    const int b = gid / DM;
    float P = 0.0f, G = -INFINITY;
    for (int c = 0; c < NCHUNK; ++c) {
        const size_t o = ((size_t)b * NCHUNK + c) * DM + d;
        Wc[o] = G + P;                      // -inf for c==0
        G = logaddexpf(G, Vc[o] - P);
        P += Ac[o];
    }
}

// =====================================================================
// Kernel 4: replay chunks, combine with prefix, emit output.
// =====================================================================
__global__ __launch_bounds__(256)
void mingru_chunk_out(const float* __restrict__ logc,
                      const float* __restrict__ logv,
                      const float* __restrict__ Wc,
                      float* __restrict__ out)
{
    const int gid = blockIdx.x * 256 + threadIdx.x;
    const int d = gid & (DM - 1);
    const int c = (gid / DM) & (NCHUNK - 1);
    const int b = gid / (DM * NCHUNK);
    const size_t base = ((size_t)b * SLEN + (size_t)c * CHUNK) * DM + d;
    const float Wv = Wc[gid];

    float la = 0.0f, m = -INFINITY, s = 0.0f;
    for (int t = 0; t < CHUNK; ++t) {
        __builtin_prefetch(&logc[base + (size_t)(t + 4) * DM], 0, 1);
        __builtin_prefetch(&logv[base + (size_t)(t + 4) * DM], 0, 1);
        const float lc = logc[base + (size_t)t * DM];
        const float lv = logv[base + (size_t)t * DM];
        la += lc;
        const float v = lv - la;
        if (v > m) { s = s * __expf(m - v) + 1.0f; m = v; }
        else       { s += __expf(v - m); }
        const float l   = m + __logf(s);
        const float lse = logaddexpf(Wv, l);
        out[base + (size_t)t * DM] = __expf(la + lse);
    }
}

// =====================================================================
extern "C" void kernel_launch(void* const* d_in, const int* in_sizes, int n_in,
                              void* d_out, int out_size, void* d_ws, size_t ws_size,
                              hipStream_t stream)
{
    const float* x = (const float*)d_in[0];        // [B, S, D] fp32
    const float* W = (const float*)d_in[1];        // [2D, D]   fp32
    float* out = (float*)d_out;                    // [B, S, D] fp32

    // workspace layout
    char* ws = (char*)d_ws;
    const size_t xb_bytes = (size_t)MTOT * DM * 2;               // 32 MB
    const size_t wb_bytes = (size_t)2 * DM * DM * 2;             //  4 MB
    const size_t plane    = (size_t)MTOT * DM * sizeof(float);   // 64 MB
    const size_t summ     = (size_t)BSZ * NCHUNK * DM * sizeof(float);
    unsigned short* xb = (unsigned short*)(ws);
    unsigned short* wb = (unsigned short*)(ws + xb_bytes);
    float* logc = (float*)(ws + xb_bytes + wb_bytes);
    float* logv = (float*)(ws + xb_bytes + wb_bytes + plane);
    float* Ac   = (float*)(ws + xb_bytes + wb_bytes + 2 * plane);
    float* Vc   = (float*)(ws + xb_bytes + wb_bytes + 2 * plane + summ);
    float* Wc   = (float*)(ws + xb_bytes + wb_bytes + 2 * plane + 2 * summ);

    cvt_f32_to_bf16<<<(MTOT * DM / 8) / 256, 256, 0, stream>>>(x, xb);
    cvt_f32_to_bf16<<<(2 * DM * DM / 8) / 256, 256, 0, stream>>>(W, wb);

    dim3 gg(MTOT / BM, DM / BN);                   // 128 x 16 blocks
    mingru_gemm_epilogue<<<gg, 256, 0, stream>>>(xb, wb, logc, logv);

    const int nchan = BSZ * NCHUNK * DM;           // 131072 streams
    mingru_chunk_scan<<<nchan / 256, 256, 0, stream>>>(logc, logv, Ac, Vc);
    mingru_chunk_prefix<<<(BSZ * DM) / 256, 256, 0, stream>>>(Ac, Vc, Wc);
    mingru_chunk_out<<<nchan / 256, 256, 0, stream>>>(logc, logv, Wc, out);
}
